// MultiHeadDiffAttention_22857815949662
// MI455X (gfx1250) — matrix operand order repaired and checked by reading, not compiled, not run on hardware
//
#include <hip/hip_runtime.h>
#include <math.h>

typedef __attribute__((ext_vector_type(16))) __bf16        v16bf;
typedef __attribute__((ext_vector_type(8)))  float         v8f;
typedef __attribute__((ext_vector_type(8)))  unsigned int  v8u;

// ---------------- helpers ----------------

__device__ __forceinline__ unsigned short bf16bits(float f) {
  unsigned int u = __builtin_bit_cast(unsigned int, f);
  u += 0x7FFFu + ((u >> 16) & 1u);           // round-to-nearest-even
  return (unsigned short)(u >> 16);
}
__device__ __forceinline__ unsigned int pack_bf2(float a, float b) {
  return (unsigned int)bf16bits(a) | ((unsigned int)bf16bits(b) << 16);
}
__device__ __forceinline__ unsigned int ld32u(const unsigned short* p) {
  return *(const unsigned int*)p;
}
// low 32 bits of a generic LDS pointer = LDS byte address
__device__ __forceinline__ unsigned lds_addr32(const void* p) {
  return (unsigned)(unsigned long long)p;
}
// async direct global->LDS copy of 16 bytes (tracked by ASYNCcnt)
__device__ __forceinline__ void async_ld_b128(unsigned lds_off,
                                              const void* gaddr) {
  asm volatile("global_load_async_to_lds_b128 %0, %1, off"
               :: "v"(lds_off), "v"((unsigned long long)gaddr)
               : "memory");
}

// A-matrix fragment (16x32 bf16, M x K): lane holds row m=lane&15.
// lanes 0-15: K = {kb..kb+7} U {16+kb..16+kb+7} with kb=0; lanes 16-31: kb=8.
__device__ __forceinline__ v16bf frag_a(const unsigned short* row, int cbase, int kb) {
  v8u u;
#pragma unroll
  for (int j = 0; j < 4; ++j) {
    u[j]     = ld32u(row + cbase + kb + 2 * j);
    u[j + 4] = ld32u(row + cbase + 16 + kb + 2 * j);
  }
  return __builtin_bit_cast(v16bf, u);
}

// B-matrix fragment (32x16 bf16, K x N): lane holds column n=lane&15,
// K = {hi*8..hi*8+7} U {16+hi*8..16+hi*8+7} (row = column of B, K contiguous).
__device__ __forceinline__ v16bf frag_b(const unsigned short* row, int cbase, int hi) {
  v8u u;
#pragma unroll
  for (int j = 0; j < 4; ++j) {
    u[j]     = ld32u(row + cbase + hi * 8 + 2 * j);
    u[j + 4] = ld32u(row + cbase + 16 + hi * 8 + 2 * j);
  }
  return __builtin_bit_cast(v16bf, u);
}

__device__ __forceinline__ v8f wmma_bf16(v16bf a, v16bf b, v8f c) {
  return __builtin_amdgcn_wmma_f32_16x16x32_bf16(false, a, false, b, (short)0, c,
                                                 false, false);
}

// ---------------- f32 -> bf16 pre-conversion (pairs) ----------------

__global__ __launch_bounds__(256)
void f32_to_bf16_kernel(const float* __restrict__ src,
                        unsigned short* __restrict__ dst, int npairs) {
  int i = blockIdx.x * 256 + threadIdx.x;
  if (i < npairs) {
    float2 f = ((const float2*)src)[i];
    ((unsigned int*)dst)[i] = pack_bf2(f.x, f.y);
  }
}

// ---------------- GEMM: C[M,N] = A[M,K] x B[K,N] (bf16 WMMA, f32 accum) ------
// Double-buffered LDS; A tile streamed via async global->LDS, B tile staged
// with an on-the-fly transpose for the B-fragment gather layout.

#define BM 128
#define BN 128
#define BK 32
#define LDA 40   // row stride (elems): 80B -> 16B-aligned b128 gathers, conflict-free
#define LDB 40

template <bool OUTBF>
__global__ __launch_bounds__(256, 1)
void gemm_wmma_kernel(const unsigned short* __restrict__ A,
                      const unsigned short* __restrict__ B,
                      void* __restrict__ Cp, int M, int N, int K) {
  __shared__ unsigned short As[2][BM][LDA];   // [m][k]
  __shared__ unsigned short Bs[2][BN][LDB];   // [n][k] (transposed on stage)

  const int m0 = blockIdx.y * BM;
  const int n0 = blockIdx.x * BN;
  const int tid  = threadIdx.x;
  const int wave = tid >> 5, lane = tid & 31;
  const int lr = lane & 15, hi = lane >> 4, kb = hi * 8;
  const int wm = (wave >> 2) * 64;   // 2 waves along M
  const int wn = (wave & 3) * 32;    // 4 waves along N

  const int arow = tid >> 1;                       // A stage: row (2 chunks/thread)
  const int bkr = tid >> 3, bns = (tid & 7) * 16;  // B stage: k-row, n-segment

  v8f acc[4][2];
  v8f zero = {0.f, 0.f, 0.f, 0.f, 0.f, 0.f, 0.f, 0.f};
#pragma unroll
  for (int i = 0; i < 4; ++i)
#pragma unroll
    for (int j = 0; j < 2; ++j) acc[i][j] = zero;

  auto stageA = [&](int kt, int buf) {
#pragma unroll
    for (int j = 0; j < 2; ++j) {
      int cq = (tid & 1) * 2 + j;   // chunk-in-row 0..3 (8 bf16 each)
      const unsigned short* g = A + (size_t)(m0 + arow) * K + kt * BK + cq * 8;
      async_ld_b128(lds_addr32(&As[buf][arow][cq * 8]), g);
    }
  };
  auto stageB = [&](int kt, int buf) {
    union { uint4 q[2]; unsigned short s[16]; } tb;
    const uint4* p = (const uint4*)(B + (size_t)(kt * BK + bkr) * N + n0 + bns);
    tb.q[0] = p[0]; tb.q[1] = p[1];
#pragma unroll
    for (int j = 0; j < 16; ++j) Bs[buf][bns + j][bkr] = tb.s[j];
  };

  const int NT = K / BK;
  stageA(0, 0);
  stageB(0, 0);

  for (int kt = 0; kt < NT; ++kt) {
    const int cur = kt & 1, nxt = cur ^ 1;
    if (kt + 1 < NT) {
      stageA(kt + 1, nxt);     // prefetch next tile while current computes
      stageB(kt + 1, nxt);
      asm volatile("s_wait_asynccnt 0x2" ::: "memory");  // current tile landed
    } else {
      asm volatile("s_wait_asynccnt 0x0" ::: "memory");
    }
    __syncthreads();

    v16bf af[4], bfr[2];
#pragma unroll
    for (int tm = 0; tm < 4; ++tm)
      af[tm] = frag_a(&As[cur][wm + tm * 16 + lr][0], 0, kb);
#pragma unroll
    for (int tn = 0; tn < 2; ++tn)
      bfr[tn] = frag_b(&Bs[cur][wn + tn * 16 + lr][0], 0, hi);
#pragma unroll
    for (int tm = 0; tm < 4; ++tm)
#pragma unroll
      for (int tn = 0; tn < 2; ++tn)
        acc[tm][tn] = wmma_bf16(af[tm], bfr[tn], acc[tm][tn]);
    __syncthreads();
  }

  // ---- writeback (C layout: lanes 0-15 m=r, lanes 16-31 m=8+r; n=lane&15) ----
#pragma unroll
  for (int tm = 0; tm < 4; ++tm)
#pragma unroll
    for (int tn = 0; tn < 2; ++tn)
#pragma unroll
      for (int r = 0; r < 8; ++r) {
        int gm = m0 + wm + tm * 16 + hi * 8 + r;
        int gn = n0 + wn + tn * 16 + lr;
        float v = acc[tm][tn][r];
        if constexpr (OUTBF)
          ((unsigned short*)Cp)[(size_t)gm * N + gn] = bf16bits(v);
        else
          ((float*)Cp)[(size_t)gm * N + gn] = v;
      }
}

// ---------------- lambda scalar ----------------

__global__ void lambda_kernel(const float* lq1, const float* lk1,
                              const float* lq2, const float* lk2, float* lam) {
  __shared__ float a1[64], a2[64];
  int i = threadIdx.x;
  a1[i] = lq1[i] * lk1[i];
  a2[i] = lq2[i] * lk2[i];
  __syncthreads();
  if (i == 0) {
    float s1 = 0.f, s2 = 0.f;
    for (int j = 0; j < 64; ++j) { s1 += a1[j]; s2 += a2[j]; }
    float LI = 0.8f - 0.6f * __expf(-0.3f);
    *lam = __expf(s1) - __expf(s2) + LI;
  }
}

// ---------------- fused dual flash-attention + diff + RMSNorm ----------------
// qkv: bf16 [B*T, 6144] = q1|k1|q2|k2|v.  One block = (b, h, 64-row q tile),
// 4 waves x 16 q-rows. Writes diff bf16 [B*T, 2048] (head-interleaved v-dims).
// Diagonal key block is peeled: only it pays the causal-mask VALU cost.

__global__ __launch_bounds__(128, 1)
void diffattn_kernel(const unsigned short* __restrict__ qkv,
                     unsigned short* __restrict__ diffo,
                     const float* __restrict__ lamp) {
  __shared__ unsigned short K1s[64][72];      // [key][d]   async-staged
  __shared__ unsigned short K2s[64][72];      // [key][d]   async-staged
  __shared__ unsigned short Vts[128][72];     // [vdim][key] (transposed)
  __shared__ unsigned short Ps[4][16][72];    // per-wave P tile [qrow][key]

  const int qb = blockIdx.x & 31;
  const int h  = (blockIdx.x >> 5) & 15;
  const int b  = (int)(blockIdx.x >> 9);

  const int tid  = threadIdx.x;
  const int wave = tid >> 5, lane = tid & 31;
  const int lr = lane & 15, hi = lane >> 4, kb = hi * 8;

  const float lam    = lamp[0];
  const float LI     = 0.8f - 0.6f * __expf(-0.3f);
  const float oscale = 1.0f - LI;

  // q fragments for this wave's 16 rows (A layout, 2 K-chunks over d=64)
  const int tq = qb * 64 + wave * 16 + lr;
  const unsigned short* qrow = qkv + ((size_t)(b * 2048 + tq)) * 6144;
  v16bf aq1[2], aq2[2];
#pragma unroll
  for (int c = 0; c < 2; ++c) {
    aq1[c] = frag_a(qrow + h * 64,        c * 32, kb);
    aq2[c] = frag_a(qrow + 2048 + h * 64, c * 32, kb);
  }

  v8f O1[8], O2[8];
  float rm1[8], rm2[8], rs1[8], rs2[8];
  v8f zero = {0.f, 0.f, 0.f, 0.f, 0.f, 0.f, 0.f, 0.f};
#pragma unroll
  for (int i = 0; i < 8; ++i) {
    O1[i] = zero; O2[i] = zero;
    rm1[i] = -3.0e38f; rm2[i] = -3.0e38f;
    rs1[i] = 0.f; rs2[i] = 0.f;
  }

  auto process_block = [&](int kblk, bool domask) {
    // ---- stage K1/K2 via async global->LDS; V transposed, key-pair packed ----
    {
      const int kr = tid >> 1, half = tid & 1;
      const unsigned short* krow =
          qkv + ((size_t)(b * 2048 + kblk * 64 + kr)) * 6144;
#pragma unroll
      for (int j = 0; j < 4; ++j) {
        int cq = half * 4 + j;  // chunk-in-row 0..7 (8 bf16 each)
        async_ld_b128(lds_addr32(&K1s[kr][cq * 8]), krow + 1024 + h * 64 + cq * 8);
        async_ld_b128(lds_addr32(&K2s[kr][cq * 8]), krow + 3072 + h * 64 + cq * 8);
      }
      {  // V: 2 adjacent keys x 32 vdims per thread -> packed dword stores
        const int kp = tid >> 2;           // key pair index 0..31
        const int vseg = (tid & 3) * 32;   // vdim segment
        const unsigned short* v0 =
            qkv + ((size_t)(b * 2048 + kblk * 64 + 2 * kp)) * 6144 +
            4096 + h * 128 + vseg;
        union { uint4 q[4]; unsigned short s[32]; } r0, r1;
#pragma unroll
        for (int j = 0; j < 4; ++j) {
          r0.q[j] = ((const uint4*)v0)[j];
          r1.q[j] = ((const uint4*)(v0 + 6144))[j];
        }
#pragma unroll
        for (int j = 0; j < 32; ++j) {
          unsigned pv = (unsigned)r0.s[j] | ((unsigned)r1.s[j] << 16);
          *(unsigned int*)&Vts[vseg + j][2 * kp] = pv;
        }
      }
      asm volatile("s_wait_asynccnt 0x0" ::: "memory");
    }
    __syncthreads();

#pragma unroll
    for (int brn = 0; brn < 2; ++brn) {
      unsigned short (*Ks)[72] = brn ? K2s : K1s;
      v8f*   O  = brn ? O2 : O1;
      float* rm = brn ? rm2 : rm1;
      float* rs = brn ? rs2 : rs1;
      v16bf* aq = brn ? aq2 : aq1;

      // S = q . k^T  (B frag = rows of K tile: column n = key, K-dim = d)
      v8f S[4];
#pragma unroll
      for (int tn = 0; tn < 4; ++tn) S[tn] = zero;
#pragma unroll
      for (int c = 0; c < 2; ++c)
#pragma unroll
        for (int tn = 0; tn < 4; ++tn) {
          v16bf bk = frag_b(&Ks[tn * 16 + lr][0], c * 32, hi);
          S[tn] = wmma_bf16(aq[c], bk, S[tn]);
        }

      // scale (+ causal mask only on the diagonal block)
      const float sscale = 0.125f;  // 1/sqrt(64)
      const int qi0 = qb * 64 + wave * 16 + hi * 8;
#pragma unroll
      for (int tn = 0; tn < 4; ++tn) {
        const int key = kblk * 64 + tn * 16 + lr;
#pragma unroll
        for (int r = 0; r < 8; ++r) {
          float sv = S[tn][r] * sscale;
          if (domask && key > qi0 + r) sv = -3.0e38f;
          S[tn][r] = sv;
        }
      }

      // online softmax: row reductions across the 16-lane half
#pragma unroll
      for (int r = 0; r < 8; ++r) {
        float mx = fmaxf(fmaxf(S[0][r], S[1][r]), fmaxf(S[2][r], S[3][r]));
#pragma unroll
        for (int off = 1; off < 16; off <<= 1)
          mx = fmaxf(mx, __shfl_xor(mx, off, 16));
        const float mnew = fmaxf(rm[r], mx);
        const float corr = __expf(rm[r] - mnew);
        rm[r] = mnew;
        float ps = 0.f;
#pragma unroll
        for (int tn = 0; tn < 4; ++tn) {
          float p = __expf(S[tn][r] - mnew);
          S[tn][r] = p;
          ps += p;
        }
#pragma unroll
        for (int off = 1; off < 16; off <<= 1)
          ps += __shfl_xor(ps, off, 16);
        rs[r] = rs[r] * corr + ps;
#pragma unroll
        for (int tn = 0; tn < 8; ++tn) O[tn][r] *= corr;
      }

      // P: C layout -> A layout via per-wave LDS tile
#pragma unroll
      for (int tn = 0; tn < 4; ++tn)
#pragma unroll
        for (int r = 0; r < 8; ++r)
          Ps[wave][hi * 8 + r][tn * 16 + lr] = bf16bits(S[tn][r]);
      asm volatile("s_wait_dscnt 0x0" ::: "memory");  // wave-local LDS RAW

      // O += P . V   (B frag from transposed V: column n = vdim, K = key)
#pragma unroll
      for (int c = 0; c < 2; ++c) {
        v16bf ap = frag_a(&Ps[wave][lr][0], c * 32, kb);
#pragma unroll
        for (int tn = 0; tn < 8; ++tn) {
          v16bf bv = frag_b(&Vts[tn * 16 + lr][0], c * 32, hi);
          O[tn] = wmma_bf16(ap, bv, O[tn]);
        }
      }
    }
    __syncthreads();
  };

  for (int kblk = 0; kblk < qb; ++kblk) process_block(kblk, false);
  process_block(qb, true);  // diagonal: only block that pays the mask cost

  // ---- finalize: diff = A1 - lam*A2, RMSNorm over 128 v-dims, scale ----
#pragma unroll
  for (int r = 0; r < 8; ++r) {
    const float i1 = 1.0f / rs1[r];
    const float i2 = lam / rs2[r];
    float dv[8];
    float ss = 0.f;
#pragma unroll
    for (int tn = 0; tn < 8; ++tn) {
      float d = O1[tn][r] * i1 - O2[tn][r] * i2;
      dv[tn] = d;
      ss += d * d;
    }
#pragma unroll
    for (int off = 1; off < 16; off <<= 1)
      ss += __shfl_xor(ss, off, 16);
    const float sc = oscale * rsqrtf(ss * (1.0f / 128.0f) + 1e-5f);
    const int trow = qb * 64 + wave * 16 + hi * 8 + r;
    const size_t ob = ((size_t)(b * 2048 + trow)) * 2048 + (size_t)h * 128;
#pragma unroll
    for (int tn = 0; tn < 8; ++tn)
      diffo[ob + tn * 16 + lr] = bf16bits(dv[tn] * sc);
  }
}

// ---------------- launch ----------------

extern "C" void kernel_launch(void* const* d_in, const int* in_sizes, int n_in,
                              void* d_out, int out_size, void* d_ws, size_t ws_size,
                              hipStream_t stream) {
  (void)in_sizes; (void)n_in; (void)out_size; (void)ws_size;
  const float* x     = (const float*)d_in[0];
  const float* Wqkv  = (const float*)d_in[1];
  const float* Wproj = (const float*)d_in[2];
  const float* lq1   = (const float*)d_in[3];
  const float* lk1   = (const float*)d_in[4];
  const float* lq2   = (const float*)d_in[5];
  const float* lk2   = (const float*)d_in[6];
  float* out = (float*)d_out;

  char* ws = (char*)d_ws;
  float* lam = (float*)ws;
  size_t o = 256;
  unsigned short* qkvb   = (unsigned short*)(ws + o); o += (size_t)4096 * 6144 * 2;
  unsigned short* diffb  = (unsigned short*)(ws + o); o += (size_t)4096 * 2048 * 2;
  unsigned short* xb     = (unsigned short*)(ws + o); o += (size_t)4096 * 2048 * 2;
  unsigned short* wqkvb  = (unsigned short*)(ws + o); o += (size_t)2048 * 6144 * 2;
  unsigned short* wprojb = (unsigned short*)(ws + o);

  lambda_kernel<<<1, 64, 0, stream>>>(lq1, lk1, lq2, lk2, lam);

  // one-shot bf16 conversions (pairs per thread)
  f32_to_bf16_kernel<<<(4096 * 2048 / 2 + 255) / 256, 256, 0, stream>>>(
      x, xb, 4096 * 2048 / 2);
  f32_to_bf16_kernel<<<(2048 * 6144 / 2 + 255) / 256, 256, 0, stream>>>(
      Wqkv, wqkvb, 2048 * 6144 / 2);
  f32_to_bf16_kernel<<<(2048 * 2048 / 2 + 255) / 256, 256, 0, stream>>>(
      Wproj, wprojb, 2048 * 2048 / 2);

  // qkv = x @ Wqkv   (bf16 in, bf16 out)   M=4096 N=6144 K=2048
  gemm_wmma_kernel<true>
      <<<dim3(6144 / BN, 4096 / BM), 256, 0, stream>>>(
          xb, wqkvb, (void*)qkvb, 4096, 6144, 2048);

  // fused dual flash-attention + diff + RMSNorm: 2*16*32 = 1024 blocks
  diffattn_kernel<<<dim3(1024), 128, 0, stream>>>(qkvb, diffb, lam);

  // out = diff @ Wproj   (bf16 in, f32 out)   M=4096 N=2048 K=2048
  gemm_wmma_kernel<false>
      <<<dim3(2048 / BN, 4096 / BM), 256, 0, stream>>>(
          diffb, wprojb, (void*)out, 4096, 2048, 2048);
}
